// MetaTwistorLNN_64098091925872
// MI455X (gfx1250) — matrix operand order repaired and checked by reading, not compiled, not run on hardware
//
#include <hip/hip_runtime.h>
#include <math.h>

// ---- problem dims (match reference) ----
constexpr int   T_STEPS = 512;
constexpr int   BATCH   = 1024;
constexpr int   IN_DIM  = 64;
constexpr int   H_DIM   = 256;
constexpr int   OUT_DIM = 32;
constexpr float DT      = 0.1f;

// ---- launch shape ----
constexpr int BM      = 16;   // batch rows per workgroup = one WMMA M-tile
constexpr int THREADS = 256;  // 8 wave32 per workgroup

typedef __attribute__((ext_vector_type(16))) __bf16 v16bf;
typedef __attribute__((ext_vector_type(8)))  float  v8f;

union Frag16 { v16bf v; uint4 q[2]; };

// ---- LDS layout (dynamic shared, ~316 KB of the 320 KB WGP LDS) ----
constexpr int OFF_WZ   = 0;                          // bf16 [256][256]  W_z   (row n, col k)
constexpr int OFF_WTAU = OFF_WZ   + H_DIM * H_DIM * 2; // bf16 [256][256]  W_tau
constexpr int OFF_ZR   = OFF_WTAU + H_DIM * H_DIM * 2; // f32  [16][256]   z_r state
constexpr int OFF_ZI   = OFF_ZR   + BM * H_DIM * 4;    // f32  [16][256]   z_i state
constexpr int OFF_TR   = OFF_ZI   + BM * H_DIM * 4;    // bf16 [16][256]   tanh(z_r)
constexpr int OFF_TI   = OFF_TR   + BM * H_DIM * 2;    // bf16 [16][256]   tanh(z_i)
constexpr int OFF_ZM   = OFF_TI   + BM * H_DIM * 2;    // bf16 [16][256]   |z|
constexpr int OFF_X    = OFF_ZM   + BM * H_DIM * 2;    // bf16 [16][64]    x_t tile
constexpr int OFF_BZ   = OFF_X    + BM * IN_DIM * 2;   // f32  [256]       b_z
constexpr int OFF_BX   = OFF_BZ   + H_DIM * 4;         // f32  [256]       b_x
constexpr int OFF_BO   = OFF_BX   + H_DIM * 4;         // f32  [32]        b_out
constexpr int SMEM_BYTES = OFF_BO + OUT_DIM * 4;       // = 323712 bytes

// ---------- math helpers ----------
__device__ __forceinline__ float fast_tanh(float x) {
  x = fminf(fmaxf(x, -15.f), 15.f);
  float e = __expf(2.f * x);
  return (e - 1.f) / (e + 1.f);
}
__device__ __forceinline__ float fast_sigmoid(float x) {
  return 1.f / (1.f + __expf(-x));
}

// ---------- WMMA ----------
__device__ __forceinline__ v8f wmma_bf16(v16bf a, v16bf b, v8f c) {
  // D(f32 16x16) = A(16x32 bf16) * B(32x16 bf16) + C
  return __builtin_amdgcn_wmma_f32_16x16x32_bf16(false, a, false, b, (short)0, c,
                                                 false, false);
}

// A-fragment (16x32 bf16) from an LDS bf16 row-major [16][rowlen] array.
// Lane L holds row m = L&15; regs 0-7 = K chunk [c0, c0+16)
// with c0 = kt*32 + 16*(L/16)  (contiguous, matches B fragment order).
__device__ __forceinline__ v16bf load_a_lds(const __bf16* st, int rowlen, int lane, int kt) {
  const int m  = lane & 15;
  const int c0 = kt * 32 + ((lane >> 4) << 4);
  const __bf16* p = st + m * rowlen + c0;
  Frag16 f;
  f.q[0] = *(const uint4*)(p);
  f.q[1] = *(const uint4*)(p + 8);
  return f.v;
}

// A-fragment built from the f32 state array (convert on the fly).
__device__ __forceinline__ v16bf load_a_lds_f32(const float* st, int lane, int kt) {
  const int m  = lane & 15;
  const int c0 = kt * 32 + ((lane >> 4) << 4);
  const float* p = st + m * H_DIM + c0;
  float4 f0 = *(const float4*)(p);
  float4 f1 = *(const float4*)(p + 4);
  float4 f2 = *(const float4*)(p + 8);
  float4 f3 = *(const float4*)(p + 12);
  v16bf v;
  v[0]  = (__bf16)f0.x; v[1]  = (__bf16)f0.y; v[2]  = (__bf16)f0.z; v[3]  = (__bf16)f0.w;
  v[4]  = (__bf16)f1.x; v[5]  = (__bf16)f1.y; v[6]  = (__bf16)f1.z; v[7]  = (__bf16)f1.w;
  v[8]  = (__bf16)f2.x; v[9]  = (__bf16)f2.y; v[10] = (__bf16)f2.z; v[11] = (__bf16)f2.w;
  v[12] = (__bf16)f3.x; v[13] = (__bf16)f3.y; v[14] = (__bf16)f3.z; v[15] = (__bf16)f3.w;
  return v;
}

// B-fragment (32x16 bf16) from an LDS bf16 weight stored row-major [n][k] (k-rowlen = 256).
// B[k][n] = W[n][k]; lane L covers column n = nt*16 + (L&15), K chunk [kb, kb+16),
// kb = kt*32 + 16*(L/16)  -> one contiguous 32-byte read.
__device__ __forceinline__ v16bf load_b_lds(const __bf16* w, int nt, int lane, int kt) {
  const int n  = nt * 16 + (lane & 15);
  const int kb = kt * 32 + ((lane >> 4) << 4);
  const __bf16* p = w + n * H_DIM + kb;
  Frag16 f;
  f.q[0] = *(const uint4*)(p);
  f.q[1] = *(const uint4*)(p + 8);
  return f.v;
}

// B-fragment from a global f32 weight [n][k] (L2-hot small weights: W_x, W_out).
__device__ __forceinline__ v16bf load_b_glb(const float* __restrict__ w, int nt, int ktot,
                                            int lane, int kt) {
  const int n  = nt * 16 + (lane & 15);
  const int kb = kt * 32 + ((lane >> 4) << 4);
  const float* p = w + n * ktot + kb;
  v16bf v;
#pragma unroll
  for (int j = 0; j < 4; ++j) {
    float4 f = *(const float4*)(p + 4 * j);
    v[4 * j + 0] = (__bf16)f.x;
    v[4 * j + 1] = (__bf16)f.y;
    v[4 * j + 2] = (__bf16)f.z;
    v[4 * j + 3] = (__bf16)f.w;
  }
  return v;
}

__global__ __launch_bounds__(THREADS, 1)
void lnn_scan_kernel(const float* __restrict__ x,
                     const float* __restrict__ Wz,
                     const float* __restrict__ Wx,
                     const float* __restrict__ Wout,
                     const float* __restrict__ Wtau,
                     const float* __restrict__ bz,
                     const float* __restrict__ bx,
                     const float* __restrict__ bo,
                     float* __restrict__ y) {
  extern __shared__ __align__(16) char smem[];
  __bf16* sWz   = (__bf16*)(smem + OFF_WZ);
  __bf16* sWtau = (__bf16*)(smem + OFF_WTAU);
  float*  sZr   = (float*)(smem + OFF_ZR);
  float*  sZi   = (float*)(smem + OFF_ZI);
  __bf16* sTr   = (__bf16*)(smem + OFF_TR);
  __bf16* sTi   = (__bf16*)(smem + OFF_TI);
  __bf16* sZm   = (__bf16*)(smem + OFF_ZM);
  __bf16* sX    = (__bf16*)(smem + OFF_X);
  float*  sBz   = (float*)(smem + OFF_BZ);
  float*  sBx   = (float*)(smem + OFF_BX);
  float*  sBo   = (float*)(smem + OFF_BO);

  const int tid  = threadIdx.x;
  const int lane = tid & 31;
  const int wid  = tid >> 5;
  const int m0   = blockIdx.x * BM;  // this WG's batch rows [m0, m0+16)

  // ---- one-time: stage W_z / W_tau as bf16 in LDS, load biases, zero state ----
  for (int i = tid; i < H_DIM * H_DIM; i += THREADS) {
    sWz[i]   = (__bf16)Wz[i];
    sWtau[i] = (__bf16)Wtau[i];
  }
  for (int i = tid; i < BM * H_DIM; i += THREADS) {
    sZr[i] = 0.f;
    sZi[i] = 0.f;
    sTr[i] = (__bf16)0.f;  // tanh(0) = 0
    sTi[i] = (__bf16)0.f;
    sZm[i] = (__bf16)0.f;  // |0| = 0
  }
  if (tid < H_DIM) { sBz[tid] = bz[tid]; sBx[tid] = bx[tid]; }
  if (tid < OUT_DIM) sBo[tid] = bo[tid];

  // ---- sequential scan over T; all heavy GEMMs hit the WMMA pipe ----
  for (int t = 0; t < T_STEPS; ++t) {
    // stage x_t tile (contiguous 16x64 f32 block -> bf16)
    const float* xt = x + ((size_t)t * BATCH + m0) * IN_DIM;
    for (int i = tid; i < BM * IN_DIM; i += THREADS) sX[i] = (__bf16)xt[i];
    __syncthreads();  // x tile + tanh/|z| staging ready; prev W_out GEMM done

    for (int q = 0; q < 2; ++q) {
      const int nt = wid * 2 + q;  // hidden N-tile owned by this wave

      // tanh(z_r)@Wz^T and tanh(z_i)@Wz^T share B-fragments
      v8f ar = {};
      v8f ai = {};
#pragma unroll
      for (int kt = 0; kt < H_DIM / 32; ++kt) {
        v16bf b  = load_b_lds(sWz, nt, lane, kt);
        v16bf aR = load_a_lds(sTr, H_DIM, lane, kt);
        v16bf aI = load_a_lds(sTi, H_DIM, lane, kt);
        ar = wmma_bf16(aR, b, ar);
        ai = wmma_bf16(aI, b, ai);
      }
      // + x_t @ Wx^T into the real part (K = 64)
#pragma unroll
      for (int kt = 0; kt < IN_DIM / 32; ++kt) {
        v16bf b  = load_b_glb(Wx, nt, IN_DIM, lane, kt);
        v16bf aX = load_a_lds(sX, IN_DIM, lane, kt);
        ar = wmma_bf16(aX, b, ar);
      }
      // |z| @ Wtau^T for the adaptive time constant
      v8f at = {};
#pragma unroll
      for (int kt = 0; kt < H_DIM / 32; ++kt) {
        v16bf b  = load_b_lds(sWtau, nt, lane, kt);
        v16bf aM = load_a_lds(sZm, H_DIM, lane, kt);
        at = wmma_bf16(aM, b, at);
      }

      // elementwise state update on the 16x16 tile this wave exclusively owns
      const int   n   = nt * 16 + (lane & 15);
      const int   mhi = (lane >> 4) << 3;
      const float bzx = sBz[n] + sBx[n];
      const float bzz = sBz[n];
#pragma unroll
      for (int r = 0; r < 8; ++r) {
        const int idx = (mhi + r) * H_DIM + n;
        float zr  = sZr[idx];
        float zi  = sZi[idx];
        float tau = fast_sigmoid(at[r]) + 1e-6f;
        float inv = 1.f / tau;
        float dzr = (-zr + ar[r] + bzx) * inv;
        float dzi = (-zi + ai[r] + bzz) * inv;
        dzr = fminf(fmaxf(dzr, -10.f), 10.f);
        dzi = fminf(fmaxf(dzi, -10.f), 10.f);
        zr += DT * dzr;
        zi += DT * dzi;
        sZr[idx] = zr;
        sZi[idx] = zi;
        // fuse next step's A-operand staging (transcendentals done exactly once)
        sTr[idx] = (__bf16)fast_tanh(zr);
        sTi[idx] = (__bf16)fast_tanh(zi);
        sZm[idx] = (__bf16)sqrtf(zr * zr + zi * zi);
      }
    }
    __syncthreads();  // new z_r visible to the output GEMM

    // y_t = z_r_new @ Wout^T + b_out  (N = 32 -> waves 0 and 1)
    if (wid < 2) {
      const int nt = wid;
      v8f acc = {};
#pragma unroll
      for (int kt = 0; kt < H_DIM / 32; ++kt) {
        v16bf b = load_b_glb(Wout, nt, H_DIM, lane, kt);
        v16bf a = load_a_lds_f32(sZr, lane, kt);
        acc = wmma_bf16(a, b, acc);
      }
      const int   n   = nt * 16 + (lane & 15);
      const int   mhi = (lane >> 4) << 3;
      const float bb  = sBo[n];
      float* yt = y + ((size_t)t * BATCH + m0) * OUT_DIM;
#pragma unroll
      for (int r = 0; r < 8; ++r) {
        yt[(mhi + r) * OUT_DIM + n] = acc[r] + bb;
      }
    }
  }
}

extern "C" void kernel_launch(void* const* d_in, const int* in_sizes, int n_in,
                              void* d_out, int out_size, void* d_ws, size_t ws_size,
                              hipStream_t stream) {
  (void)in_sizes; (void)n_in; (void)out_size; (void)d_ws; (void)ws_size;
  const float* x    = (const float*)d_in[0];
  const float* Wz   = (const float*)d_in[1];
  const float* Wx   = (const float*)d_in[2];
  const float* Wout = (const float*)d_in[3];
  const float* Wtau = (const float*)d_in[4];
  const float* bz   = (const float*)d_in[5];
  const float* bx   = (const float*)d_in[6];
  const float* bo   = (const float*)d_in[7];
  float* y = (float*)d_out;

  // 64 persistent workgroups (one 16-row batch tile each), 8 wave32 per WG,
  // ~316 KB dynamic LDS per WG (W_z + W_tau resident as bf16 for all 512 steps).
  dim3 grid(BATCH / BM);
  dim3 block(THREADS);
  lnn_scan_kernel<<<grid, block, SMEM_BYTES, stream>>>(x, Wz, Wx, Wout, Wtau, bz, bx, bo, y);
}
